// TraceUnifiedModelV3_63385127354392
// MI455X (gfx1250) — compile-verified, matrix operand order checked by reading, not run on hardware
//
#include <hip/hip_runtime.h>
#include <hip/hip_bf16.h>

// ---------------------------------------------------------------------------
// Types for CDNA5 WMMA (wave32): v_wmma_f32_16x16x32_bf16
// ---------------------------------------------------------------------------
typedef __attribute__((ext_vector_type(16))) __bf16 v16bf;
typedef __attribute__((ext_vector_type(8)))  float  v8f;

union Frag16 {
    v16bf v;
    unsigned short u[16];
};

__device__ __forceinline__ unsigned short f2bf(float f) {
    union { float f; unsigned int u; } x; x.f = f;
    unsigned int u = x.u;
    unsigned int r = u + 0x7FFFu + ((u >> 16) & 1u);   // round-to-nearest-even
    return (unsigned short)(r >> 16);
}

__device__ __forceinline__ float sigm(float x) { return 1.0f / (1.0f + expf(-x)); }

static inline int cdiv(long long a, long long b) { return (int)((a + b - 1) / b); }

// ---------------------------------------------------------------------------
// Utility kernels
// ---------------------------------------------------------------------------
__global__ void zero_f32(float* p, long long n) {
    long long t = (long long)blockIdx.x * blockDim.x + threadIdx.x;
    if (t < n) p[t] = 0.0f;
}
__global__ void zero_i32(int* p, long long n) {
    long long t = (long long)blockIdx.x * blockDim.x + threadIdx.x;
    if (t < n) p[t] = 0;
}
__global__ void cvt_f32_bf16(const float* __restrict__ s, unsigned short* __restrict__ d, long long n) {
    long long t = (long long)blockIdx.x * blockDim.x + threadIdx.x;
    if (t < n) d[t] = f2bf(s[t]);
}

// ---------------------------------------------------------------------------
// Generic WMMA GEMM:  C[M,Nout] = act( A[M,K](f32) @ W[Nout,K](bf16)^T + bias )
// Optional device-side row gather/scatter via (rowIdx, lvlOff, lvl):
//   rows of A/C are rowIdx[lvlOff[lvl] .. lvlOff[lvl+1]).
// Block = 128 threads = 4 waves; each wave owns 32 rows x 64 cols
// (2 A fragments x 4 B fragments = 8 WMMAs per K-step).
// Requirements: K % 32 == 0, Nout % 64 == 0 (all call sites satisfy this).
// ---------------------------------------------------------------------------
__global__ __launch_bounds__(128)
void gemm_wmma_bf16(const float* __restrict__ A, int lda,
                    const unsigned short* __restrict__ Wbf, // [Nout,K]
                    const float* __restrict__ bias,
                    float* __restrict__ C, int ldc,
                    int M, int K, int Nout, int relu,
                    const int* __restrict__ rowIdx,
                    const int* __restrict__ lvlOff, int lvl)
{
    const int wave = threadIdx.x >> 5;
    const int lane = threadIdx.x & 31;
    const int r    = lane & 15;
    const bool hi  = lane >= 16;

    int listBase = 0, cnt = M;
    if (rowIdx) {
        listBase = lvlOff[lvl];
        cnt = lvlOff[lvl + 1] - listBase;
    }
    const int mBase = blockIdx.x * 128 + wave * 32;
    if (mBase >= cnt) return;                  // wave-uniform: EXEC stays full
    const int nBase = blockIdx.y * 64;

    // A rows for this lane's two fragments (both lane halves hold the same rows)
    const float* Arow[2];
    #pragma unroll
    for (int mt = 0; mt < 2; ++mt) {
        int mpos = mBase + mt * 16 + r;
        int arow;
        if (rowIdx) arow = rowIdx[listBase + ((mpos < cnt) ? mpos : (cnt - 1))];
        else        arow = (mpos < M) ? mpos : (M - 1);
        Arow[mt] = A + (size_t)arow * lda + (hi ? 8 : 0);
    }
    const unsigned short* Wbase = Wbf + (size_t)(nBase + r) * K + (hi ? 8 : 0);

    v8f acc[2][4] = {};

    for (int k0 = 0; k0 < K; k0 += 32) {
        // --- A fragments: lane r<16 holds K {0..7,16..23}, hi half {8..15,24..31}
        Frag16 a[2];
        #pragma unroll
        for (int mt = 0; mt < 2; ++mt) {
            const float* p = Arow[mt] + k0;
            float4 f0 = *(const float4*)(p);
            float4 f1 = *(const float4*)(p + 4);
            float4 f2 = *(const float4*)(p + 16);
            float4 f3 = *(const float4*)(p + 20);
            a[mt].u[0]  = f2bf(f0.x); a[mt].u[1]  = f2bf(f0.y);
            a[mt].u[2]  = f2bf(f0.z); a[mt].u[3]  = f2bf(f0.w);
            a[mt].u[4]  = f2bf(f1.x); a[mt].u[5]  = f2bf(f1.y);
            a[mt].u[6]  = f2bf(f1.z); a[mt].u[7]  = f2bf(f1.w);
            a[mt].u[8]  = f2bf(f2.x); a[mt].u[9]  = f2bf(f2.y);
            a[mt].u[10] = f2bf(f2.z); a[mt].u[11] = f2bf(f2.w);
            a[mt].u[12] = f2bf(f3.x); a[mt].u[13] = f2bf(f3.y);
            a[mt].u[14] = f2bf(f3.z); a[mt].u[15] = f2bf(f3.w);
        }
        #pragma unroll
        for (int nt = 0; nt < 4; ++nt) {
            // column (nBase+nt*16+r) of B == row of W
            const unsigned short* q = Wbase + (size_t)(nt * 16) * K + k0;
            Frag16 b;
            *(uint4*)&b.u[0] = *(const uint4*)(q);
            *(uint4*)&b.u[8] = *(const uint4*)(q + 16);
            acc[0][nt] = __builtin_amdgcn_wmma_f32_16x16x32_bf16(
                false, a[0].v, false, b.v, (short)0, acc[0][nt], false, false);
            acc[1][nt] = __builtin_amdgcn_wmma_f32_16x16x32_bf16(
                false, a[1].v, false, b.v, (short)0, acc[1][nt], false, false);
        }
    }

    // --- Store C: VGPR rr -> row rr (+8 for hi half), col = lane&15
    #pragma unroll
    for (int mt = 0; mt < 2; ++mt) {
        #pragma unroll
        for (int nt = 0; nt < 4; ++nt) {
            const int col = nBase + nt * 16 + r;
            const float bv = bias ? bias[col] : 0.0f;
            #pragma unroll
            for (int rr = 0; rr < 8; ++rr) {
                const int mLoc = mt * 16 + rr + (hi ? 8 : 0);
                const int pos  = mBase + mLoc;
                if (pos >= cnt) continue;
                const int crow = rowIdx ? rowIdx[listBase + pos] : pos;
                float v = acc[mt][nt][rr] + bv;
                if (relu) v = v > 0.0f ? v : 0.0f;
                C[(size_t)crow * ldc + col] = v;
            }
        }
    }
}

// ---------------------------------------------------------------------------
// Stage 0: embeddings + latency MLP -> xcat[N,384]
// ---------------------------------------------------------------------------
__global__ void build_xcat(const int* __restrict__ api, const int* __restrict__ status,
                           const int* __restrict__ node, const int* __restrict__ depth,
                           const int* __restrict__ pos,  const float* __restrict__ lat,
                           const float* __restrict__ e_api, const float* __restrict__ e_st,
                           const float* __restrict__ e_nd,  const float* __restrict__ e_dp,
                           const float* __restrict__ e_ps,
                           const float* __restrict__ lw1, const float* __restrict__ lb1,
                           const float* __restrict__ lw2, const float* __restrict__ lb2,
                           float* __restrict__ xcat, int N)
{
    long long t = (long long)blockIdx.x * blockDim.x + threadIdx.x;
    if (t >= (long long)N * 384) return;
    int n = (int)(t / 384), j = (int)(t % 384);
    int seg = j >> 6, jj = j & 63;
    float v;
    if (seg == 0)      v = e_api[api[n] * 64 + jj];
    else if (seg == 1) v = e_st[status[n] * 64 + jj];
    else if (seg == 2) v = e_nd[node[n] * 64 + jj];
    else if (seg == 3) { int d = depth[n]; d = d < 0 ? 0 : (d > 63 ? 63 : d); v = e_dp[d * 64 + jj]; }
    else if (seg == 4) { int p = pos[n];   p = p < 0 ? 0 : (p > 2047 ? 2047 : p); v = e_ps[p * 64 + jj]; }
    else {
        float l = lat[n], s = 0.0f;
        for (int m = 0; m < 64; ++m) {
            float h = l * lw1[m] + lb1[m];
            h = h > 0.0f ? h : 0.0f;
            s += h * lw2[jj * 64 + m];
        }
        v = s + lb2[jj];
    }
    xcat[(size_t)n * 384 + j] = v;
}

// ---------------------------------------------------------------------------
// GCN helpers
// ---------------------------------------------------------------------------
__global__ void edge_degrees(const int* __restrict__ src, const int* __restrict__ dst, int E,
                             float* __restrict__ degOut, float* __restrict__ degIn)
{
    int e = blockIdx.x * blockDim.x + threadIdx.x;
    if (e >= E) return;
    atomicAdd(&degOut[src[e]], 1.0f);
    atomicAdd(&degIn[dst[e]], 1.0f);
}

__global__ void gcn_scatter(const float* __restrict__ x, const int* __restrict__ src,
                            const int* __restrict__ dst, const float* __restrict__ degOut,
                            float* __restrict__ agg, int E)
{
    long long t = (long long)blockIdx.x * blockDim.x + threadIdx.x;
    int e = (int)(t >> 5);
    int c = ((int)t & 31) * 4;
    if (e >= E) return;
    int s = src[e], d = dst[e];
    float doo = degOut[s];
    float ns = doo > 0.0f ? rsqrtf(doo) : 0.0f;
    float4 v = *(const float4*)(x + (size_t)s * 128 + c);
    float* ar = agg + (size_t)d * 128 + c;
    atomicAdd(ar + 0, v.x * ns);
    atomicAdd(ar + 1, v.y * ns);
    atomicAdd(ar + 2, v.z * ns);
    atomicAdd(ar + 3, v.w * ns);
}

__global__ void gcn_scale(const float* __restrict__ agg, const float* __restrict__ degIn,
                          float* __restrict__ out, int N)
{
    long long t = (long long)blockIdx.x * blockDim.x + threadIdx.x;
    if (t >= (long long)N * 128) return;
    int n = (int)(t >> 7);
    float di = degIn[n];
    float nd = di > 0.0f ? rsqrtf(di) : 0.0f;
    out[t] = agg[t] * nd;
}

// ---------------------------------------------------------------------------
// TreeLSTM: level bucketing
// ---------------------------------------------------------------------------
__global__ void level_count(const int* __restrict__ depth, int* __restrict__ cnt, int N)
{
    int n = blockIdx.x * blockDim.x + threadIdx.x;
    if (n >= N) return;
    int d = depth[n]; d = d < 0 ? 0 : (d > 31 ? 31 : d);
    atomicAdd(&cnt[d], 1);
}
__global__ void level_scan(const int* __restrict__ cnt, int* __restrict__ off)
{
    if (blockIdx.x == 0 && threadIdx.x == 0) {
        int s = 0;
        for (int i = 0; i < 32; ++i) { off[i] = s; s += cnt[i]; }
        off[32] = s;
    }
}
__global__ void level_fill(const int* __restrict__ depth, const int* __restrict__ off,
                           int* __restrict__ cur, int* __restrict__ nodes, int N)
{
    int n = blockIdx.x * blockDim.x + threadIdx.x;
    if (n >= N) return;
    int d = depth[n]; d = d < 0 ? 0 : (d > 31 ? 31 : d);
    int p = atomicAdd(&cur[d], 1);
    nodes[off[d] + p] = n;
}

// Per-level elementwise LSTM cell update (writes h into cat[:,256:384]).
__global__ void tree_update(const int* __restrict__ lvlNodes, const int* __restrict__ lvlOff, int lvl,
                            const float* __restrict__ iou_x, const float* __restrict__ iouh_t,
                            const float* __restrict__ fc_acc, const float* __restrict__ nchild,
                            const float* __restrict__ x0, float* __restrict__ cat,
                            float* __restrict__ c_state)
{
    int base = lvlOff[lvl];
    int cnt  = lvlOff[lvl + 1] - base;
    long long t = (long long)blockIdx.x * blockDim.x + threadIdx.x;
    int i = (int)(t >> 7), j = (int)t & 127;
    if (i >= cnt) return;
    int n = lvlNodes[base + i];
    size_t r3 = (size_t)n * 384, r1 = (size_t)n * 128;
    float ig = iou_x[r3 + j]       + iouh_t[r3 + j];
    float og = iou_x[r3 + 128 + j] + iouh_t[r3 + 128 + j];
    float ug = iou_x[r3 + 256 + j] + iouh_t[r3 + 256 + j];
    float c_int = fc_acc[r1 + j] + sigm(ig) * tanhf(ug);
    float h_int = sigm(og) * tanhf(c_int);
    bool hc = nchild[n] > 0.0f;
    float x = x0[r1 + j];
    cat[r3 + 256 + j] = hc ? h_int : x;
    c_state[r1 + j]   = hc ? c_int : tanhf(x);
}

// Push finalized node's (f*c, h) into its parent's accumulators.
__global__ void tree_push(const int* __restrict__ lvlNodes, const int* __restrict__ lvlOff, int lvl,
                          const int* __restrict__ parent, const float* __restrict__ fx_x,
                          const float* __restrict__ fhv, const float* __restrict__ cat,
                          const float* __restrict__ c_state,
                          float* __restrict__ fc_acc, float* __restrict__ hsum_acc)
{
    int base = lvlOff[lvl];
    int cnt  = lvlOff[lvl + 1] - base;
    long long t = (long long)blockIdx.x * blockDim.x + threadIdx.x;
    int i = (int)(t >> 7), j = (int)t & 127;
    if (i >= cnt) return;
    int n = lvlNodes[base + i];
    int p = parent[n];
    if (p < 0) return;
    float f = sigm(fx_x[(size_t)p * 128 + j] + fhv[(size_t)n * 128 + j]);
    atomicAdd(&fc_acc[(size_t)p * 128 + j], f * c_state[(size_t)n * 128 + j]);
    atomicAdd(&hsum_acc[(size_t)p * 128 + j], cat[(size_t)n * 384 + 256 + j]);
}

// ---------------------------------------------------------------------------
// Gate: logits = g1 @ gate_w2^T + b2 ; softmax ; combine -> d_out node rows
// One 128-thread block per node.
// ---------------------------------------------------------------------------
__global__ __launch_bounds__(128)
void gate_combine(const float* __restrict__ cat, const float* __restrict__ g1,
                  const float* __restrict__ gw2, const float* __restrict__ gb2,
                  float* __restrict__ out)
{
    __shared__ float red[3][128];
    __shared__ float gg[3];
    int n = blockIdx.x, t = threadIdx.x;
    float a = g1[(size_t)n * 128 + t];
    #pragma unroll
    for (int k = 0; k < 3; ++k) red[k][t] = a * gw2[k * 128 + t];
    __syncthreads();
    for (int s = 64; s > 0; s >>= 1) {
        if (t < s) {
            #pragma unroll
            for (int k = 0; k < 3; ++k) red[k][t] += red[k][t + s];
        }
        __syncthreads();
    }
    if (t == 0) {
        float l0 = red[0][0] + gb2[0], l1 = red[1][0] + gb2[1], l2 = red[2][0] + gb2[2];
        float m = fmaxf(l0, fmaxf(l1, l2));
        float e0 = expf(l0 - m), e1 = expf(l1 - m), e2 = expf(l2 - m);
        float s = e0 + e1 + e2;
        gg[0] = e0 / s; gg[1] = e1 / s; gg[2] = e2 / s;
    }
    __syncthreads();
    const float* cr = cat + (size_t)n * 384;
    out[(size_t)n * 128 + t] = gg[0] * cr[t] + gg[1] * cr[128 + t] + gg[2] * cr[256 + t];
}

// ---------------------------------------------------------------------------
// Graph means
// ---------------------------------------------------------------------------
__global__ void graph_accum(const float* __restrict__ cat, const int* __restrict__ gid,
                            float* __restrict__ parts, float* __restrict__ gcnt, int N)
{
    long long t = (long long)blockIdx.x * blockDim.x + threadIdx.x;
    if (t >= (long long)N * 384) return;
    int n = (int)(t / 384), j = (int)(t % 384);
    int g = gid[n];
    atomicAdd(&parts[(size_t)g * 384 + j], cat[(size_t)n * 384 + j]);
    if (j == 0) atomicAdd(&gcnt[g], 1.0f);
}
__global__ void parts_div(float* __restrict__ parts, const float* __restrict__ gcnt, int G)
{
    long long t = (long long)blockIdx.x * blockDim.x + threadIdx.x;
    if (t >= (long long)G * 384) return;
    int g = (int)(t / 384);
    float c = gcnt[g];
    parts[t] /= (c > 0.0f ? c : 1.0f);
}

// ---------------------------------------------------------------------------
// Host side
// ---------------------------------------------------------------------------
static inline void* wsTake(char*& p, size_t bytes) {
    void* r = (void*)p;
    p += (bytes + 255) & ~(size_t)255;
    return r;
}

extern "C" void kernel_launch(void* const* d_in, const int* in_sizes, int n_in,
                              void* d_out, int out_size, void* d_ws, size_t ws_size,
                              hipStream_t stream)
{
    const int*   api    = (const int*)d_in[0];
    const int*   status = (const int*)d_in[1];
    const int*   node   = (const int*)d_in[2];
    const int*   depth  = (const int*)d_in[3];
    const int*   pos    = (const int*)d_in[4];
    const int*   parent = (const int*)d_in[5];
    const int*   gid    = (const int*)d_in[6];
    const int*   csrc   = (const int*)d_in[7];
    const int*   cdst   = (const int*)d_in[8];
    const int*   hsrc   = (const int*)d_in[9];
    const int*   hdst   = (const int*)d_in[10];
    const float* lat    = (const float*)d_in[12];
    const float* e_api  = (const float*)d_in[13];
    const float* e_st   = (const float*)d_in[14];
    const float* e_nd   = (const float*)d_in[15];
    const float* e_dp   = (const float*)d_in[16];
    const float* e_ps   = (const float*)d_in[17];
    const float* lw1    = (const float*)d_in[18];
    const float* lb1    = (const float*)d_in[19];
    const float* lw2    = (const float*)d_in[20];
    const float* lb2    = (const float*)d_in[21];
    const float* merge_w = (const float*)d_in[22];
    const float* merge_b = (const float*)d_in[23];
    const float* call1_w = (const float*)d_in[24];
    const float* call1_b = (const float*)d_in[25];
    const float* call2_w = (const float*)d_in[26];
    const float* call2_b = (const float*)d_in[27];
    const float* host1_w = (const float*)d_in[28];
    const float* host1_b = (const float*)d_in[29];
    const float* host2_w = (const float*)d_in[30];
    const float* host2_b = (const float*)d_in[31];
    const float* ioux_w  = (const float*)d_in[32];
    const float* iouh_w  = (const float*)d_in[33];
    const float* fx_w    = (const float*)d_in[34];
    const float* fh_w    = (const float*)d_in[35];
    const float* gate_w1 = (const float*)d_in[36];
    const float* gate_b1 = (const float*)d_in[37];
    const float* gate_w2 = (const float*)d_in[38];
    const float* gate_b2 = (const float*)d_in[39];
    const float* fuse_w1 = (const float*)d_in[40];
    const float* fuse_b1 = (const float*)d_in[41];
    const float* fuse_w2 = (const float*)d_in[42];
    const float* fuse_b2 = (const float*)d_in[43];

    const int N  = in_sizes[0];
    const int Ec = in_sizes[7];
    const int Eh = in_sizes[9];
    int G = out_size / 128 - N;
    if (G < 1) G = 1;

    float* out = (float*)d_out;

    // ---- workspace layout ----
    char* wp = (char*)d_ws;
    unsigned short* merge_bf = (unsigned short*)wsTake(wp, (size_t)128 * 384 * 2);
    unsigned short* call1_bf = (unsigned short*)wsTake(wp, (size_t)128 * 128 * 2);
    unsigned short* call2_bf = (unsigned short*)wsTake(wp, (size_t)128 * 128 * 2);
    unsigned short* host1_bf = (unsigned short*)wsTake(wp, (size_t)128 * 128 * 2);
    unsigned short* host2_bf = (unsigned short*)wsTake(wp, (size_t)128 * 128 * 2);
    unsigned short* ioux_bf  = (unsigned short*)wsTake(wp, (size_t)384 * 128 * 2);
    unsigned short* iouh_bf  = (unsigned short*)wsTake(wp, (size_t)384 * 128 * 2);
    unsigned short* fx_bf    = (unsigned short*)wsTake(wp, (size_t)128 * 128 * 2);
    unsigned short* fh_bf    = (unsigned short*)wsTake(wp, (size_t)128 * 128 * 2);
    unsigned short* gate1_bf = (unsigned short*)wsTake(wp, (size_t)128 * 384 * 2);
    unsigned short* fuse1_bf = (unsigned short*)wsTake(wp, (size_t)128 * 384 * 2);
    unsigned short* fuse2_bf = (unsigned short*)wsTake(wp, (size_t)128 * 128 * 2);

    float* xcat  = (float*)wsTake(wp, (size_t)N * 384 * 4);  // later reused as iouh_tmp
    float* x0    = (float*)wsTake(wp, (size_t)N * 128 * 4);
    float* cat   = (float*)wsTake(wp, (size_t)N * 384 * 4);  // [h_call | h_host | h_tree]
    float* iou_x = (float*)wsTake(wp, (size_t)N * 384 * 4);
    float* fx_x  = (float*)wsTake(wp, (size_t)N * 128 * 4);
    float* agg   = (float*)wsTake(wp, (size_t)N * 128 * 4);
    float* a_in  = (float*)wsTake(wp, (size_t)N * 128 * 4);  // later reused as g1
    float* htmp  = (float*)wsTake(wp, (size_t)N * 128 * 4);
    float* hsum  = (float*)wsTake(wp, (size_t)N * 128 * 4);
    float* fcac  = (float*)wsTake(wp, (size_t)N * 128 * 4);
    float* fhv   = (float*)wsTake(wp, (size_t)N * 128 * 4);
    float* cst   = (float*)wsTake(wp, (size_t)N * 128 * 4);
    float* degOutC = (float*)wsTake(wp, (size_t)N * 4);      // == n_child
    float* degInC  = (float*)wsTake(wp, (size_t)N * 4);
    float* degOutH = (float*)wsTake(wp, (size_t)N * 4);
    float* degInH  = (float*)wsTake(wp, (size_t)N * 4);
    int* lvlCnt   = (int*)wsTake(wp, 32 * 4);
    int* lvlOff   = (int*)wsTake(wp, 33 * 4);
    int* lvlCur   = (int*)wsTake(wp, 32 * 4);
    int* lvlNodes = (int*)wsTake(wp, (size_t)N * 4);
    float* parts = (float*)wsTake(wp, (size_t)G * 384 * 4);
    float* gcnt  = (float*)wsTake(wp, (size_t)G * 4);
    float* f1    = (float*)wsTake(wp, (size_t)G * 128 * 4);

    float* iouh_tmp = xcat;  // xcat dead after merge GEMM
    float* g1 = a_in;        // a_in dead after GCN

    const int T = 256;

    // ---- weights -> bf16 ----
    struct { const float* s; unsigned short* d; long long n; } cv[] = {
        { merge_w, merge_bf, 128 * 384 }, { call1_w, call1_bf, 128 * 128 },
        { call2_w, call2_bf, 128 * 128 }, { host1_w, host1_bf, 128 * 128 },
        { host2_w, host2_bf, 128 * 128 }, { ioux_w, ioux_bf, 384 * 128 },
        { iouh_w, iouh_bf, 384 * 128 },   { fx_w, fx_bf, 128 * 128 },
        { fh_w, fh_bf, 128 * 128 },       { gate_w1, gate1_bf, 128 * 384 },
        { fuse_w1, fuse1_bf, 128 * 384 }, { fuse_w2, fuse2_bf, 128 * 128 },
    };
    for (auto& c : cv)
        cvt_f32_bf16<<<cdiv(c.n, T), T, 0, stream>>>(c.s, c.d, c.n);

    // ---- stage 0: features ----
    build_xcat<<<cdiv((long long)N * 384, T), T, 0, stream>>>(
        api, status, node, depth, pos, lat, e_api, e_st, e_nd, e_dp, e_ps,
        lw1, lb1, lw2, lb2, xcat, N);

    // x0 = xcat @ merge_w^T + merge_b
    gemm_wmma_bf16<<<dim3(cdiv(N, 128), 2), 128, 0, stream>>>(
        xcat, 384, merge_bf, merge_b, x0, 128, N, 384, 128, 0, nullptr, nullptr, 0);

    // TreeLSTM input projections
    gemm_wmma_bf16<<<dim3(cdiv(N, 128), 6), 128, 0, stream>>>(
        x0, 128, ioux_bf, nullptr, iou_x, 384, N, 128, 384, 0, nullptr, nullptr, 0);
    gemm_wmma_bf16<<<dim3(cdiv(N, 128), 2), 128, 0, stream>>>(
        x0, 128, fx_bf, nullptr, fx_x, 128, N, 128, 128, 0, nullptr, nullptr, 0);

    // ---- GCN call path -> cat[:, 0:128] ----
    zero_f32<<<cdiv(N, T), T, 0, stream>>>(degOutC, N);
    zero_f32<<<cdiv(N, T), T, 0, stream>>>(degInC, N);
    edge_degrees<<<cdiv(Ec, T), T, 0, stream>>>(csrc, cdst, Ec, degOutC, degInC);

    zero_f32<<<cdiv((long long)N * 128, T), T, 0, stream>>>(agg, (long long)N * 128);
    gcn_scatter<<<cdiv((long long)Ec * 32, T), T, 0, stream>>>(x0, csrc, cdst, degOutC, agg, Ec);
    gcn_scale<<<cdiv((long long)N * 128, T), T, 0, stream>>>(agg, degInC, a_in, N);
    gemm_wmma_bf16<<<dim3(cdiv(N, 128), 2), 128, 0, stream>>>(
        a_in, 128, call1_bf, call1_b, htmp, 128, N, 128, 128, 1, nullptr, nullptr, 0);

    zero_f32<<<cdiv((long long)N * 128, T), T, 0, stream>>>(agg, (long long)N * 128);
    gcn_scatter<<<cdiv((long long)Ec * 32, T), T, 0, stream>>>(htmp, csrc, cdst, degOutC, agg, Ec);
    gcn_scale<<<cdiv((long long)N * 128, T), T, 0, stream>>>(agg, degInC, a_in, N);
    gemm_wmma_bf16<<<dim3(cdiv(N, 128), 2), 128, 0, stream>>>(
        a_in, 128, call2_bf, call2_b, cat + 0, 384, N, 128, 128, 1, nullptr, nullptr, 0);

    // ---- GCN host path -> cat[:, 128:256] ----
    zero_f32<<<cdiv(N, T), T, 0, stream>>>(degOutH, N);
    zero_f32<<<cdiv(N, T), T, 0, stream>>>(degInH, N);
    edge_degrees<<<cdiv(Eh, T), T, 0, stream>>>(hsrc, hdst, Eh, degOutH, degInH);

    zero_f32<<<cdiv((long long)N * 128, T), T, 0, stream>>>(agg, (long long)N * 128);
    gcn_scatter<<<cdiv((long long)Eh * 32, T), T, 0, stream>>>(x0, hsrc, hdst, degOutH, agg, Eh);
    gcn_scale<<<cdiv((long long)N * 128, T), T, 0, stream>>>(agg, degInH, a_in, N);
    gemm_wmma_bf16<<<dim3(cdiv(N, 128), 2), 128, 0, stream>>>(
        a_in, 128, host1_bf, host1_b, htmp, 128, N, 128, 128, 1, nullptr, nullptr, 0);

    zero_f32<<<cdiv((long long)N * 128, T), T, 0, stream>>>(agg, (long long)N * 128);
    gcn_scatter<<<cdiv((long long)Eh * 32, T), T, 0, stream>>>(htmp, hsrc, hdst, degOutH, agg, Eh);
    gcn_scale<<<cdiv((long long)N * 128, T), T, 0, stream>>>(agg, degInH, a_in, N);
    gemm_wmma_bf16<<<dim3(cdiv(N, 128), 2), 128, 0, stream>>>(
        a_in, 128, host2_bf, host2_b, cat + 128, 384, N, 128, 128, 1, nullptr, nullptr, 0);

    // ---- TreeLSTM -> cat[:, 256:384] ----
    zero_f32<<<cdiv((long long)N * 128, T), T, 0, stream>>>(hsum, (long long)N * 128);
    zero_f32<<<cdiv((long long)N * 128, T), T, 0, stream>>>(fcac, (long long)N * 128);
    zero_i32<<<1, 64, 0, stream>>>(lvlCnt, 32);
    zero_i32<<<1, 64, 0, stream>>>(lvlCur, 32);
    level_count<<<cdiv(N, T), T, 0, stream>>>(depth, lvlCnt, N);
    level_scan<<<1, 1, 0, stream>>>(lvlCnt, lvlOff);
    level_fill<<<cdiv(N, T), T, 0, stream>>>(depth, lvlOff, lvlCur, lvlNodes, N);

    for (int L = 31; L >= 0; --L) {
        // iouh_tmp[rows] = hsum[rows] @ iouh_w^T   (compacted rows at this level)
        gemm_wmma_bf16<<<dim3(cdiv(N, 128), 6), 128, 0, stream>>>(
            hsum, 128, iouh_bf, nullptr, iouh_tmp, 384, N, 128, 384, 0, lvlNodes, lvlOff, L);
        tree_update<<<cdiv((long long)N * 128, T), T, 0, stream>>>(
            lvlNodes, lvlOff, L, iou_x, iouh_tmp, fcac, degOutC /*n_child*/, x0, cat, cst);
        // fhv[rows] = h[rows] @ fh_w^T   (h lives at cat[:,256:], lda=384)
        gemm_wmma_bf16<<<dim3(cdiv(N, 128), 2), 128, 0, stream>>>(
            cat + 256, 384, fh_bf, nullptr, fhv, 128, N, 128, 128, 0, lvlNodes, lvlOff, L);
        tree_push<<<cdiv((long long)N * 128, T), T, 0, stream>>>(
            lvlNodes, lvlOff, L, parent, fx_x, fhv, cat, cst, fcac, hsum);
    }

    // ---- gate + node output ----
    gemm_wmma_bf16<<<dim3(cdiv(N, 128), 2), 128, 0, stream>>>(
        cat, 384, gate1_bf, gate_b1, g1, 128, N, 384, 128, 1, nullptr, nullptr, 0);
    gate_combine<<<N, 128, 0, stream>>>(cat, g1, gate_w2, gate_b2, out);

    // ---- graph-level output ----
    zero_f32<<<cdiv((long long)G * 384, T), T, 0, stream>>>(parts, (long long)G * 384);
    zero_f32<<<cdiv(G, T), T, 0, stream>>>(gcnt, G);
    graph_accum<<<cdiv((long long)N * 384, T), T, 0, stream>>>(cat, gid, parts, gcnt, N);
    parts_div<<<cdiv((long long)G * 384, T), T, 0, stream>>>(parts, gcnt, G);

    gemm_wmma_bf16<<<dim3(cdiv(G, 128), 2), 128, 0, stream>>>(
        parts, 384, fuse1_bf, fuse_b1, f1, 128, G, 384, 128, 1, nullptr, nullptr, 0);
    gemm_wmma_bf16<<<dim3(cdiv(G, 128), 2), 128, 0, stream>>>(
        f1, 128, fuse2_bf, fuse_b2, out + (size_t)N * 128, 128, G, 128, 128, 0, nullptr, nullptr, 0);
}
